// PolicyValueEfficientGTrXL_56023553409325
// MI455X (gfx1250) — matrix operand order repaired and checked by reading, not compiled, not run on hardware
//
#include <hip/hip_runtime.h>
#include <hip/hip_bf16.h>
#include <math.h>

// ---------------------------------------------------------------------------
// GTrXL block for MI455X (gfx1250, wave32, WMMA bf16 -> f32 accumulate)
// ---------------------------------------------------------------------------

typedef __attribute__((ext_vector_type(16))) __bf16 v16bf;
typedef __attribute__((ext_vector_type(8)))  float  v8f;

#define DEV __device__ __forceinline__

constexpr int Bb   = 16;        // batch
constexpr int Tt   = 256;       // query time
constexpr int Mm   = 256;       // memory time
constexpr int Dd   = 1024;      // model dim
constexpr int Hh   = 16;        // heads
constexpr int Ss   = Mm + Tt;   // 512 kv time
constexpr int HD   = Dd / Hh;   // 64 head dim
constexpr int MLPH = 4 * Dd;    // 4096
constexpr int BT   = Bb * Tt;   // 4096 rows
constexpr int BS   = Bb * Ss;   // 8192 rows

DEV v8f wmma_bf16(v16bf a, v16bf b, v8f c) {
  // D = A(16x32 bf16) * B(32x16 bf16) + C(16x16 f32)
  return __builtin_amdgcn_wmma_f32_16x16x32_bf16(
      /*neg_a=*/false, a, /*neg_b=*/false, b,
      /*c_mod=*/(short)0, c, /*reuse_a=*/false, /*reuse_b=*/false);
}

// A-matrix fragment (16x32, bf16): lane[0..15]=rows M, halves split K.
// element e -> K = e + 16*half (matches B fragment order)
DEV v16bf frag_a_bf16(const __bf16* base, int ld, int lane) {
  const int m = lane & 15, half = lane >> 4;
  const __bf16* row = base + (size_t)m * ld;
  v16bf f;
#pragma unroll
  for (int e = 0; e < 16; e++) {
    int k  = e + (half << 4);
    f[e] = row[k];
  }
  return f;
}

DEV v16bf frag_a_f32(const float* base, int ld, int lane) {
  const int m = lane & 15, half = lane >> 4;
  const float* row = base + (size_t)m * ld;
  v16bf f;
#pragma unroll
  for (int e = 0; e < 16; e++) {
    int k  = e + (half << 4);
    f[e] = (__bf16)row[k];
  }
  return f;
}

// B-matrix fragment (32x16, bf16) from transposed tile Bt[n][k]:
// lane[0..15]=column N, element e -> K = e + 16*half.
DEV v16bf frag_b_bf16(const __bf16* base, int ld, int lane) {
  const int n = lane & 15, half = lane >> 4;
  const __bf16* row = base + (size_t)n * ld;
  v16bf f;
#pragma unroll
  for (int e = 0; e < 16; e++) f[e] = row[e + (half << 4)];
  return f;
}

DEV float half16_max(float x) {
  x = fmaxf(x, __shfl_xor(x, 1, 32));
  x = fmaxf(x, __shfl_xor(x, 2, 32));
  x = fmaxf(x, __shfl_xor(x, 4, 32));
  x = fmaxf(x, __shfl_xor(x, 8, 32));
  return x;
}
DEV float half16_sum(float x) {
  x += __shfl_xor(x, 1, 32);
  x += __shfl_xor(x, 2, 32);
  x += __shfl_xor(x, 4, 32);
  x += __shfl_xor(x, 8, 32);
  return x;
}

// ---------------------------------------------------------------------------
// Generic tiled GEMM: C[M,N] = act(A[M,K] @ W[K,N] + bias), bf16 WMMA math.
// Requires M%128==0, N%128==0, K%32==0 (true for every call site here).
// 8 wave32 waves; wave grid 2x4; each wave owns a 64x32 C sub-tile (4x2 WMMA).
// Double-buffered LDS; next tile's global b128 loads are issued before the
// current tile's WMMAs so VMEM overlaps the matrix pipe.
// ---------------------------------------------------------------------------
template <bool HASB, bool RELU, bool OUTBF>
__global__ __launch_bounds__(256)
void gemm_bf16_kernel(const float* __restrict__ A, const float* __restrict__ W,
                      const float* __restrict__ bias,
                      float* __restrict__ Cf, __bf16* __restrict__ Cbf,
                      int Nd, int Kd) {
  __shared__ __bf16 As[2][128][36];   // [buf][m][k]
  __shared__ __bf16 Bs[2][128][36];   // [buf][n][k] (transposed W tile)
  const int tid  = threadIdx.x;
  const int lane = tid & 31, wid = tid >> 5;
  const int wm = wid >> 2, wn = wid & 3;
  const size_t m0 = (size_t)blockIdx.y * 128;
  const size_t n0 = (size_t)blockIdx.x * 128;

  v8f acc[4][2] = {};
  float4 ra[4], rb[4];

  auto gload = [&](int kb) {
#pragma unroll
    for (int i = 0; i < 4; i++) {
      int li = i * 256 + tid;
      int r = li >> 3, c4 = li & 7;                 // A: 128 rows x 8 float4
      ra[i] = *(const float4*)&A[(m0 + r) * (size_t)Kd + kb + c4 * 4];
      int k = li >> 5, n4 = li & 31;                // W: 32 k x 32 float4
      rb[i] = *(const float4*)&W[(size_t)(kb + k) * Nd + n0 + n4 * 4];
    }
  };
  auto sstore = [&](int buf) {
#pragma unroll
    for (int i = 0; i < 4; i++) {
      int li = i * 256 + tid;
      int r = li >> 3, c4 = li & 7;
      __bf16* ap = &As[buf][r][c4 * 4];
      ap[0] = (__bf16)ra[i].x; ap[1] = (__bf16)ra[i].y;
      ap[2] = (__bf16)ra[i].z; ap[3] = (__bf16)ra[i].w;
      int k = li >> 5, n4 = li & 31;
      Bs[buf][n4 * 4 + 0][k] = (__bf16)rb[i].x;
      Bs[buf][n4 * 4 + 1][k] = (__bf16)rb[i].y;
      Bs[buf][n4 * 4 + 2][k] = (__bf16)rb[i].z;
      Bs[buf][n4 * 4 + 3][k] = (__bf16)rb[i].w;
    }
  };
  auto compute = [&](int buf) {
    v16bf af[4], bf2[2];
#pragma unroll
    for (int i = 0; i < 4; i++)
      af[i] = frag_a_bf16(&As[buf][wm * 64 + i * 16][0], 36, lane);
#pragma unroll
    for (int j = 0; j < 2; j++)
      bf2[j] = frag_b_bf16(&Bs[buf][wn * 32 + j * 16][0], 36, lane);
#pragma unroll
    for (int i = 0; i < 4; i++)
#pragma unroll
      for (int j = 0; j < 2; j++) acc[i][j] = wmma_bf16(af[i], bf2[j], acc[i][j]);
  };

  const int nsteps = Kd / 32;
  gload(0);
  sstore(0);
  for (int step = 0; step < nsteps; step++) {
    __syncthreads();
    if (step + 1 < nsteps) gload((step + 1) * 32);  // VMEM for next tile
    compute(step & 1);                              // WMMA on current tile
    if (step + 1 < nsteps) sstore((step + 1) & 1);  // LDS fill of other buffer
  }

  const int nl = lane & 15, half = lane >> 4;
#pragma unroll
  for (int i = 0; i < 4; i++)
#pragma unroll
    for (int j = 0; j < 2; j++) {
      size_t ng = n0 + wn * 32 + j * 16 + nl;
      float bv = HASB ? bias[ng] : 0.f;
#pragma unroll
      for (int v = 0; v < 8; v++) {
        size_t mg = m0 + wm * 64 + i * 16 + v + 8 * half;
        float xv = acc[i][j][v] + bv;
        if (RELU) xv = fmaxf(xv, 0.f);
        if (OUTBF) Cbf[mg * (size_t)Nd + ng] = (__bf16)xv;
        else       Cf [mg * (size_t)Nd + ng] = xv;
      }
    }
}

// ---------------------------------------------------------------------------
// Position scores: pos[h,t,s,b] = sum_d rk[t,s,h,d] * Q[b,t,h,d]
// One wave per (s-tile-of-16, t, h): M=s, N=b, K=64 -> two WMMAs.
// ---------------------------------------------------------------------------
__global__ __launch_bounds__(32)
void pos_score_kernel(const __bf16* __restrict__ rk, const float* __restrict__ Q,
                      float* __restrict__ pos) {
  const int s0 = blockIdx.x * 16;
  const int t  = blockIdx.y;
  const int h  = blockIdx.z;
  const int lane = threadIdx.x;
  const int nl = lane & 15, half = lane >> 4;

  const __bf16* abase = rk + ((size_t)t * Ss + s0) * Dd + h * HD;
  v16bf a0 = frag_a_bf16(abase, Dd, lane);        // d = 0..31
  v16bf a1 = frag_a_bf16(abase + 32, Dd, lane);   // d = 32..63

  v16bf b0, b1;                                   // Bt[n=b][k=d] = Q[b,t,h*64+d]
  const float* qb = Q + ((size_t)nl * Tt + t) * Dd + h * HD;
#pragma unroll
  for (int e = 0; e < 16; e++) {
    int k = e + (half << 4);
    b0[e] = (__bf16)qb[k];
    b1[e] = (__bf16)qb[32 + k];
  }

  v8f acc = {};
  acc = wmma_bf16(a0, b0, acc);
  acc = wmma_bf16(a1, b1, acc);

  float* cb = pos + (((size_t)h * Tt + t) * Ss + s0) * 16;   // [.., s, b]
#pragma unroll
  for (int v = 0; v < 8; v++) cb[(size_t)(v + 8 * half) * 16 + nl] = acc[v];
}

// ---------------------------------------------------------------------------
// Flash attention with relative-position bias. One wave per (b, h, 16 t-rows).
// Streams S in 32-wide chunks: QK^T (2 WMMAs/16x16 tile) + bias, online
// softmax (cross-lane reductions within wave32 halves), P@V (4 WMMAs).
// ---------------------------------------------------------------------------
__global__ __launch_bounds__(32)
void flash_attn_kernel(const float* __restrict__ Q, const float* __restrict__ K,
                       const float* __restrict__ V, const float* __restrict__ pos,
                       float* __restrict__ O) {
  __shared__ __bf16 Kl[32][HD + 4];  // [s][d]  (B-frag for QK^T)
  __shared__ __bf16 Vt[HD][36];      // [d][s]  (B-frag for P@V)
  __shared__ __bf16 Pl[16][36];      // [t][s]  (A-frag for P@V)

  const int t0 = blockIdx.x * 16;
  const int h  = blockIdx.y;
  const int b  = blockIdx.z;
  const int lane = threadIdx.x;
  const int nl = lane & 15, half = lane >> 4;
  const float scale = 0.125f;   // 1/sqrt(64)

  const float* qbase = Q + ((size_t)b * Tt + t0) * Dd + h * HD;
  v16bf aq0 = frag_a_f32(qbase, Dd, lane);        // d 0..31
  v16bf aq1 = frag_a_f32(qbase + 32, Dd, lane);   // d 32..63

  float rmax[8], rsum[8];
#pragma unroll
  for (int v = 0; v < 8; v++) { rmax[v] = -INFINITY; rsum[v] = 0.f; }
  v8f o[4] = {};

  for (int sc = 0; sc < Ss / 32; sc++) {
    const int s0 = sc * 32;
#pragma unroll
    for (int i = 0; i < 16; i++) {           // stage K,V chunk (32 x 64) as b128
      int li = i * 32 + lane;
      int s = li >> 4, d4 = li & 15;
      size_t gi = ((size_t)b * Ss + s0 + s) * Dd + h * HD + d4 * 4;
      float4 k4 = *(const float4*)&K[gi];
      float4 v4 = *(const float4*)&V[gi];
      __bf16* kp = &Kl[s][d4 * 4];
      kp[0] = (__bf16)k4.x; kp[1] = (__bf16)k4.y;
      kp[2] = (__bf16)k4.z; kp[3] = (__bf16)k4.w;
      Vt[d4 * 4 + 0][s] = (__bf16)v4.x;
      Vt[d4 * 4 + 1][s] = (__bf16)v4.y;
      Vt[d4 * 4 + 2][s] = (__bf16)v4.z;
      Vt[d4 * 4 + 3][s] = (__bf16)v4.w;
    }
    __syncthreads();

    float sv[2][8];
#pragma unroll
    for (int ss = 0; ss < 2; ss++) {
      v8f sacc = {};
      v16bf bk0 = frag_b_bf16(&Kl[ss * 16][0], HD + 4, lane);       // d 0..31
      v16bf bk1 = frag_b_bf16(&Kl[ss * 16][0] + 32, HD + 4, lane);  // d 32..63
      sacc = wmma_bf16(aq0, bk0, sacc);
      sacc = wmma_bf16(aq1, bk1, sacc);
#pragma unroll
      for (int v = 0; v < 8; v++) {
        int m = v + 8 * half;
        int s = s0 + ss * 16 + nl;
        float pb = pos[(((size_t)h * Tt + t0 + m) * Ss + s) * 16 + b];
        sv[ss][v] = scale * (sacc[v] + pb);
      }
    }

#pragma unroll
    for (int v = 0; v < 8; v++) {            // online softmax per row
      float cm = half16_max(fmaxf(sv[0][v], sv[1][v]));
      float nm = fmaxf(rmax[v], cm);
      float alpha = __expf(rmax[v] - nm);    // first iter: exp(-inf)=0
      rmax[v] = nm;
      float e0 = __expf(sv[0][v] - nm);
      float e1 = __expf(sv[1][v] - nm);
      rsum[v] = rsum[v] * alpha + half16_sum(e0 + e1);
#pragma unroll
      for (int j = 0; j < 4; j++) o[j][v] = o[j][v] * alpha;
      int m = v + 8 * half;
      Pl[m][nl]      = (__bf16)e0;
      Pl[m][16 + nl] = (__bf16)e1;
    }
    __syncthreads();

    v16bf ap = frag_a_bf16(&Pl[0][0], 36, lane);
#pragma unroll
    for (int j = 0; j < 4; j++) {
      v16bf bv = frag_b_bf16(&Vt[j * 16][0], 36, lane);
      o[j] = wmma_bf16(ap, bv, o[j]);
    }
    __syncthreads();
  }

#pragma unroll
  for (int j = 0; j < 4; j++)
#pragma unroll
    for (int v = 0; v < 8; v++) {
      int m = v + 8 * half;
      O[((size_t)b * Tt + t0 + m) * Dd + h * HD + j * 16 + nl] = o[j][v] / rsum[v];
    }
}

// ---------------------------------------------------------------------------
// LayerNorm over D=1024, one 256-thread block per row.
// ---------------------------------------------------------------------------
__global__ __launch_bounds__(256)
void ln_kernel(const float* __restrict__ X, const float* __restrict__ w,
               const float* __restrict__ bvec, float* __restrict__ Y) {
  __shared__ float r1[256], r2[256];
  const int row = blockIdx.x, tid = threadIdx.x;
  const float* xr = X + (size_t)row * Dd;
  float s = 0.f, s2 = 0.f;
  for (int i = tid; i < Dd; i += 256) { float v = xr[i]; s += v; s2 += v * v; }
  r1[tid] = s; r2[tid] = s2; __syncthreads();
  for (int o = 128; o; o >>= 1) {
    if (tid < o) { r1[tid] += r1[tid + o]; r2[tid] += r2[tid + o]; }
    __syncthreads();
  }
  float mu  = r1[0] * (1.f / Dd);
  float var = r2[0] * (1.f / Dd) - mu * mu;
  float rs  = rsqrtf(var + 1e-5f);
  for (int i = tid; i < Dd; i += 256)
    Y[(size_t)row * Dd + i] = (xr[i] - mu) * rs * w[i] + bvec[i];
}

// Gated residual: g = sigmoid(LN(G;gw,gb)); Y = g*branch + (1-g)*resid
__global__ __launch_bounds__(256)
void gate_apply_kernel(const float* __restrict__ G, const float* __restrict__ gw,
                       const float* __restrict__ gb, const float* __restrict__ branch,
                       const float* __restrict__ resid, float* __restrict__ Y) {
  __shared__ float r1[256], r2[256];
  const int row = blockIdx.x, tid = threadIdx.x;
  const float* g = G + (size_t)row * Dd;
  float s = 0.f, s2 = 0.f;
  for (int i = tid; i < Dd; i += 256) { float v = g[i]; s += v; s2 += v * v; }
  r1[tid] = s; r2[tid] = s2; __syncthreads();
  for (int o = 128; o; o >>= 1) {
    if (tid < o) { r1[tid] += r1[tid + o]; r2[tid] += r2[tid + o]; }
    __syncthreads();
  }
  float mu  = r1[0] * (1.f / Dd);
  float var = r2[0] * (1.f / Dd) - mu * mu;
  float rs  = rsqrtf(var + 1e-5f);
  for (int i = tid; i < Dd; i += 256) {
    float ln = (g[i] - mu) * rs * gw[i] + gb[i];
    float gg = 1.f / (1.f + __expf(-ln));
    size_t idx = (size_t)row * Dd + i;
    Y[idx] = gg * branch[idx] + (1.f - gg) * resid[idx];
  }
}

// kv[b,s,:] = s<M ? mem[b,s,:] : xn[b,s-M,:]
__global__ void build_kv_kernel(const float* __restrict__ mem,
                                const float* __restrict__ xn,
                                float* __restrict__ kv) {
  size_t idx = (size_t)blockIdx.x * blockDim.x + threadIdx.x;
  if (idx >= (size_t)BS * Dd) return;
  int d = (int)(idx % Dd);
  size_t rr = idx / Dd;
  int s = (int)(rr % Ss);
  int b = (int)(rr / Ss);
  kv[idx] = (s < Mm) ? mem[((size_t)b * Mm + s) * Dd + d]
                     : xn[((size_t)b * Tt + (s - Mm)) * Dd + d];
}

// cat[r, 0:D] = a[r, :]; cat[r, D:2D] = c2[r, :]
__global__ void build_cat_kernel(const float* __restrict__ a,
                                 const float* __restrict__ c2,
                                 float* __restrict__ cat) {
  size_t idx = (size_t)blockIdx.x * blockDim.x + threadIdx.x;
  if (idx >= (size_t)BT * 2 * Dd) return;
  int c = (int)(idx % (2 * Dd));
  size_t r = idx / (2 * Dd);
  cat[idx] = (c < Dd) ? a[r * Dd + c] : c2[r * Dd + (c - Dd)];
}

// ---------------------------------------------------------------------------
extern "C" void kernel_launch(void* const* d_in, const int* in_sizes, int n_in,
                              void* d_out, int out_size, void* d_ws, size_t ws_size,
                              hipStream_t stream) {
  (void)in_sizes; (void)n_in; (void)out_size; (void)ws_size;
  const float* x    = (const float*)d_in[0];
  const float* mem  = (const float*)d_in[1];
  const float* relp = (const float*)d_in[2];
  const float* anw  = (const float*)d_in[3];
  const float* anb  = (const float*)d_in[4];
  const float* Wq   = (const float*)d_in[5];
  const float* Wk   = (const float*)d_in[6];
  const float* Wv   = (const float*)d_in[7];
  const float* Wo   = (const float*)d_in[8];
  const float* bo   = (const float*)d_in[9];
  const float* Wr   = (const float*)d_in[10];
  const float* agw  = (const float*)d_in[11];
  const float* agnw = (const float*)d_in[12];
  const float* agnb = (const float*)d_in[13];
  const float* mnw  = (const float*)d_in[14];
  const float* mnb  = (const float*)d_in[15];
  const float* W1   = (const float*)d_in[16];
  const float* b1   = (const float*)d_in[17];
  const float* W2   = (const float*)d_in[18];
  const float* b2   = (const float*)d_in[19];
  const float* mgw  = (const float*)d_in[20];
  const float* mgnw = (const float*)d_in[21];
  const float* mgnb = (const float*)d_in[22];
  float* out = (float*)d_out;

  char* ws = (char*)d_ws;
  size_t off = 0;
  auto arena = [&](size_t bytes) -> char* {
    char* p = ws + off;
    off += (bytes + 255) & ~(size_t)255;
    return p;
  };
  float*  xn   = (float*)arena((size_t)BT * Dd * 4);          // also reused as xn2
  float*  kv   = (float*)arena((size_t)BS * Dd * 4);
  float*  Qb   = (float*)arena((size_t)BT * Dd * 4);
  float*  Kb   = (float*)arena((size_t)BS * Dd * 4);
  float*  Vb   = (float*)arena((size_t)BS * Dd * 4);
  float*  attn = (float*)arena((size_t)BT * Dd * 4);
  float*  proj = (float*)arena((size_t)BT * Dd * 4);
  float*  cat  = (float*)arena((size_t)BT * 2 * Dd * 4);      // reused for both gates
  float*  g1   = (float*)arena((size_t)BT * Dd * 4);          // reused for both gates
  float*  x1   = (float*)arena((size_t)BT * Dd * 4);
  float*  hbuf = (float*)arena((size_t)BT * MLPH * 4);
  float*  mbuf = (float*)arena((size_t)BT * Dd * 4);
  float*  pos  = (float*)arena((size_t)Hh * Tt * Ss * 16 * 4);
  __bf16* rk   = (__bf16*)arena((size_t)Tt * Ss * Dd * 2);

  // ---- attention branch ----
  ln_kernel<<<BT, 256, 0, stream>>>(x, anw, anb, xn);
  {
    size_t tot = (size_t)BS * Dd;
    build_kv_kernel<<<(unsigned)((tot + 255) / 256), 256, 0, stream>>>(mem, xn, kv);
  }
  gemm_bf16_kernel<false, false, false><<<dim3(Dd / 128, BT / 128), 256, 0, stream>>>(
      xn, Wq, nullptr, Qb, nullptr, Dd, Dd);
  gemm_bf16_kernel<false, false, false><<<dim3(Dd / 128, BS / 128), 256, 0, stream>>>(
      kv, Wk, nullptr, Kb, nullptr, Dd, Dd);
  gemm_bf16_kernel<false, false, false><<<dim3(Dd / 128, BS / 128), 256, 0, stream>>>(
      kv, Wv, nullptr, Vb, nullptr, Dd, Dd);
  // rk = rel_pos @ Wr  (131072 x 1024 x 1024), bf16 output feeds WMMA stage
  gemm_bf16_kernel<false, false, true><<<dim3(Dd / 128, (Tt * Ss) / 128), 256, 0, stream>>>(
      relp, Wr, nullptr, nullptr, rk, Dd, Dd);
  pos_score_kernel<<<dim3(Ss / 16, Tt, Hh), 32, 0, stream>>>(rk, Qb, pos);
  flash_attn_kernel<<<dim3(Tt / 16, Hh, Bb), 32, 0, stream>>>(Qb, Kb, Vb, pos, attn);
  gemm_bf16_kernel<true, false, false><<<dim3(Dd / 128, BT / 128), 256, 0, stream>>>(
      attn, Wo, bo, proj, nullptr, Dd, Dd);

  // ---- attention gated residual ----
  {
    size_t tot = (size_t)BT * 2 * Dd;
    build_cat_kernel<<<(unsigned)((tot + 255) / 256), 256, 0, stream>>>(x, proj, cat);
  }
  gemm_bf16_kernel<false, false, false><<<dim3(Dd / 128, BT / 128), 256, 0, stream>>>(
      cat, agw, nullptr, g1, nullptr, Dd, 2 * Dd);
  gate_apply_kernel<<<BT, 256, 0, stream>>>(g1, agnw, agnb, proj, x, x1);

  // ---- MLP branch ----
  ln_kernel<<<BT, 256, 0, stream>>>(x1, mnw, mnb, xn);   // xn reused as xn2
  gemm_bf16_kernel<true, true, false><<<dim3(MLPH / 128, BT / 128), 256, 0, stream>>>(
      xn, W1, b1, hbuf, nullptr, MLPH, Dd);
  gemm_bf16_kernel<true, false, false><<<dim3(Dd / 128, BT / 128), 256, 0, stream>>>(
      hbuf, W2, b2, mbuf, nullptr, Dd, MLPH);

  // ---- MLP gated residual ----
  {
    size_t tot = (size_t)BT * 2 * Dd;
    build_cat_kernel<<<(unsigned)((tot + 255) / 256), 256, 0, stream>>>(x1, mbuf, cat);
  }
  gemm_bf16_kernel<false, false, false><<<dim3(Dd / 128, BT / 128), 256, 0, stream>>>(
      cat, mgw, nullptr, g1, nullptr, Dd, 2 * Dd);
  gate_apply_kernel<<<BT, 256, 0, stream>>>(g1, mgnw, mgnb, mbuf, x1, out);
}
